// Attention_57097295233561
// MI455X (gfx1250) — matrix operand order repaired and checked by reading, not compiled, not run on hardware
//
#include <hip/hip_runtime.h>

// Problem constants (from reference)
#define BD   2
#define SD   2048
#define DIMD 2048
#define NHD  16
#define HDD  128
#define MD   (BD * SD)   // 4096 rows of the flattened activation matrix

typedef __attribute__((ext_vector_type(16))) __bf16 v16bf;
typedef __attribute__((ext_vector_type(8)))  __bf16 v8bf;
typedef __attribute__((ext_vector_type(8)))  float  v8f;
typedef __attribute__((ext_vector_type(4)))  unsigned int v4u;
typedef __attribute__((ext_vector_type(8)))  int v8i;
typedef __attribute__((ext_vector_type(4)))  int v4i;

union AFrag { v16bf v; v8bf h[2]; };

__device__ __forceinline__ v8f wmma_bf16(v16bf a, v16bf b, v8f c) {
  // (neg_a, A, neg_b, B, c_mod, C, reuse_a, reuse_b)
  return __builtin_amdgcn_wmma_f32_16x16x32_bf16(false, a, false, b, (short)0, c,
                                                 false, false);
}

// ---------------------------------------------------------------------------
// TDM: DMA a (rows x 32) bf16 tile (row stride = kdim elements) into LDS with
// 16B padding after each 64B row (LDS row stride becomes 80B -> conflict-free
// B-fragment reads: bank start n*20 mod 64 is distinct per lane).
// Descriptor per CDNA5 ISA sec 8.3/8.4; 6-arg builtin (clang-23 form).
// ---------------------------------------------------------------------------
__device__ __forceinline__ void tdm_load_tile(unsigned lds_addr, const void* gptr,
                                              unsigned rows, unsigned kdim) {
  unsigned long long ga = (unsigned long long)(size_t)gptr;
  v4u g0;
  g0.x = 1u;                                            // count=1 (valid), user mode
  g0.y = lds_addr;                                      // LDS byte address
  g0.z = (unsigned)(ga & 0xFFFFFFFFu);                  // global_addr[31:0]
  g0.w = (unsigned)((ga >> 32) & 0x1FFFFFFu) | (2u << 30);  // addr[56:32] | type=2
  const unsigned td0 = kdim;     // tensor dim0 (elements)
  const unsigned td1 = rows;     // tensor dim1
  const unsigned long long st0 = kdim;   // dim0 stride (elements)
  v8i g1;
  // workgroup_mask=0 | data_size=1(2B)<<16 | pad_enable<<20 |
  // pad_interval=3(16 DW)<<22 | pad_amount=3(4 DW)<<25
  g1[0] = (int)(0x00010000u | (1u << 20) | (3u << 22) | (3u << 25));
  g1[1] = (int)((td0 & 0xFFFFu) << 16);                           // dim0[15:0]
  g1[2] = (int)(((td0 >> 16) & 0xFFFFu) | ((td1 & 0xFFFFu) << 16));
  g1[3] = (int)(((td1 >> 16) & 0xFFFFu) | (32u << 16));           // tile_dim0=32
  g1[4] = (int)(rows & 0xFFFFu);                                  // tile_dim1, tile_dim2=0
  g1[5] = (int)(st0 & 0xFFFFFFFFu);                               // stride0[31:0]
  g1[6] = (int)((st0 >> 32) & 0xFFFFu);                           // stride0[47:32], stride1=0
  g1[7] = 0;
  v4i z4 = {0, 0, 0, 0};
  v8i z8 = {0, 0, 0, 0, 0, 0, 0, 0};
  __builtin_amdgcn_tensor_load_to_lds(g0, g1, z4, z4, z8, 0);
}

// ---------------------------------------------------------------------------
// f32 -> bf16 cast
// ---------------------------------------------------------------------------
__global__ __launch_bounds__(256) void cast_f32_to_bf16(const float* __restrict__ in,
                                                        __bf16* __restrict__ out,
                                                        size_t n) {
  size_t i = (size_t)blockIdx.x * 256 + threadIdx.x;
  if (i < n) out[i] = (__bf16)in[i];
}

// ---------------------------------------------------------------------------
// GEMM: out[m,n] = sum_k A[m,k] * W[n,k]  (A: MxK bf16, W: NxK bf16, rowmajor)
// Block = 256 threads / 8 waves computes a 128x64 tile; TDM double-buffers the
// 128x32 A tile and 64x32 W tile into LDS (80B padded row stride); each wave
// computes a 32x32 register tile: 4 WMMAs per K-step from 128B of LDS.
// MODE 0: bf16 out[m*N+n]  MODE 1: bf16 out[n*M+m]  MODE 2: f32 out[m*N+n]
// ---------------------------------------------------------------------------
template <int MODE>
__global__ __launch_bounds__(256) void gemm_bf16_tdm(const __bf16* __restrict__ A,
                                                     const __bf16* __restrict__ W,
                                                     void* __restrict__ out,
                                                     int Mdim, int Ndim, int Kdim) {
  // per buffer: A 128 rows * 80B + B 64 rows * 80B = 15360B; double buffered
  __shared__ __align__(128) char smem[2 * (128 * 80 + 64 * 80)];
  const int lane = threadIdx.x & 31;
  const int wav  = threadIdx.x >> 5;
  const int m0 = blockIdx.y * 128;
  const int n0 = blockIdx.x * 64;
  const int wm = (wav & 3) * 32;     // wave row offset inside block tile
  const int wn = (wav >> 2) * 32;    // wave col offset inside block tile
  const int am = lane & 15;
  const int kh = lane >> 4;
  const unsigned ldsBase  = (unsigned)(size_t)(&smem[0]);
  const unsigned bufBytes = 128 * 80 + 64 * 80;

  v8f acc[2][2] = {};
  const int nk = Kdim / 32;

  if (wav == 0) {  // prologue: kick off buffer 0
    tdm_load_tile(ldsBase,            A + (size_t)m0 * Kdim, 128, Kdim);
    tdm_load_tile(ldsBase + 128 * 80, W + (size_t)n0 * Kdim,  64, Kdim);
  }

  for (int ki = 0; ki < nk; ++ki) {
    const int db = ki & 1;
    __syncthreads();                       // buffer db^1 is free to overwrite
    if (wav == 0) {
      if (ki + 1 < nk) {
        const unsigned nb = ldsBase + (unsigned)(db ^ 1) * bufBytes;
        tdm_load_tile(nb,            A + (size_t)m0 * Kdim + (ki + 1) * 32, 128, Kdim);
        tdm_load_tile(nb + 128 * 80, W + (size_t)n0 * Kdim + (ki + 1) * 32,  64, Kdim);
        __builtin_amdgcn_s_wait_tensorcnt((short)2);  // in-order: current pair done
      } else {
        __builtin_amdgcn_s_wait_tensorcnt((short)0);
      }
    }
    __syncthreads();                       // buffer db visible to all waves

    const char* base = smem + (size_t)db * bufBytes;
    const char* aB = base;
    const char* bB = base + 128 * 80;
    AFrag a0, a1, b0, b1;
    // A frag (16x32): lane row = am; chunks at bytes {kh*16, 32+kh*16}
    a0.h[0] = *(const v8bf*)(aB + (wm + am) * 80 + kh * 16);
    a0.h[1] = *(const v8bf*)(aB + (wm + am) * 80 + 32 + kh * 16);
    a1.h[0] = *(const v8bf*)(aB + (wm + 16 + am) * 80 + kh * 16);
    a1.h[1] = *(const v8bf*)(aB + (wm + 16 + am) * 80 + 32 + kh * 16);
    // B frag (32x16): lane col = am; chunks at bytes {kh*16, 32+kh*16}
    b0.h[0] = *(const v8bf*)(bB + (wn + am) * 80 + kh * 16);
    b0.h[1] = *(const v8bf*)(bB + (wn + am) * 80 + 32 + kh * 16);
    b1.h[0] = *(const v8bf*)(bB + (wn + 16 + am) * 80 + kh * 16);
    b1.h[1] = *(const v8bf*)(bB + (wn + 16 + am) * 80 + 32 + kh * 16);

    acc[0][0] = wmma_bf16(a0.v, b0.v, acc[0][0]);
    acc[0][1] = wmma_bf16(a0.v, b1.v, acc[0][1]);
    acc[1][0] = wmma_bf16(a1.v, b0.v, acc[1][0]);
    acc[1][1] = wmma_bf16(a1.v, b1.v, acc[1][1]);
  }

  // D layout: lane n = lane&15, VGPR r -> row r + 8*kh
#pragma unroll
  for (int i = 0; i < 2; ++i) {
#pragma unroll
    for (int j = 0; j < 2; ++j) {
#pragma unroll
      for (int r = 0; r < 8; ++r) {
        int m = m0 + wm + i * 16 + r + 8 * kh;
        int n = n0 + wn + j * 16 + am;
        if (MODE == 0)
          ((__bf16*)out)[(size_t)m * Ndim + n] = (__bf16)acc[i][j][r];
        else if (MODE == 1)
          ((__bf16*)out)[(size_t)n * Mdim + m] = (__bf16)acc[i][j][r];
        else
          ((float*)out)[(size_t)m * Ndim + n] = acc[i][j][r];
      }
    }
  }
}

// ---------------------------------------------------------------------------
// RoPE (in place on bf16 Q and K), pairs (2j, 2j+1) within each head dim
// ---------------------------------------------------------------------------
__global__ __launch_bounds__(256) void rope_kernel(__bf16* __restrict__ q,
                                                   __bf16* __restrict__ k,
                                                   const float* __restrict__ fc) {
  const size_t total = (size_t)MD * NHD * (HDD / 2);
  size_t i = (size_t)blockIdx.x * 256 + threadIdx.x;
  if (i >= total) return;
  const int half = HDD / 2;                   // 64
  const int perRow = NHD * half;              // 1024
  int m  = (int)(i / perRow);
  int rem = (int)(i % perRow);
  int hh = rem / half;
  int j  = rem % half;
  int s  = m & (SD - 1);
  float c  = fc[((size_t)s * half + j) * 2 + 0];
  float sn = fc[((size_t)s * half + j) * 2 + 1];
  size_t base = (size_t)m * DIMD + hh * HDD + 2 * j;
  float qr = (float)q[base], qi = (float)q[base + 1];
  q[base]     = (__bf16)(qr * c - qi * sn);
  q[base + 1] = (__bf16)(qr * sn + qi * c);
  float kr = (float)k[base], ki = (float)k[base + 1];
  k[base]     = (__bf16)(kr * c - ki * sn);
  k[base + 1] = (__bf16)(kr * sn + ki * c);
}

// ---------------------------------------------------------------------------
// Causal flash attention, one wave per 16-query tile.
// Q,K: bf16 [m, e] (e = h*HD + hd);  Vt: bf16 [e, m];  O: bf16 [m, e]
// ---------------------------------------------------------------------------
__global__ __launch_bounds__(256) void flash_attn_wmma(const __bf16* __restrict__ Q,
                                                       const __bf16* __restrict__ Km,
                                                       const __bf16* __restrict__ Vt,
                                                       __bf16* __restrict__ O) {
  __shared__ __bf16 pshared[8][16 * 32];      // per-wave P staging (1KB each)
  const int lane = threadIdx.x & 31;
  const int wav  = threadIdx.x >> 5;
  const int b = blockIdx.x >> 4;
  const int h = blockIdx.x & 15;
  const int qt = blockIdx.y * 8 + wav;        // query tile index, 0..127
  const int am = lane & 15;
  const int kh = lane >> 4;
  const float scale = 0.0883883476483184f;    // 1/sqrt(128)

  // Q A-fragments for the 4 HD chunks of 32
  const __bf16* qp = Q + (size_t)(b * SD + qt * 16 + am) * DIMD + h * HDD;
  AFrag qf[4];
#pragma unroll
  for (int c = 0; c < 4; ++c) {
    qf[c].h[0] = *reinterpret_cast<const v8bf*>(qp + c * 32 + kh * 16);
    qf[c].h[1] = *reinterpret_cast<const v8bf*>(qp + c * 32 + kh * 16 + 8);
  }

  v8f o[8] = {};                              // 16x128 output accumulator
  float mrow[8], lrow[8];
#pragma unroll
  for (int r = 0; r < 8; ++r) { mrow[r] = -1e30f; lrow[r] = 0.f; }

  const int qhi = qt * 16 + 15;
  __bf16* pw = &pshared[wav][0];

  for (int kk = 0; kk <= qhi; kk += 32) {
    // scores: two 16x16 tiles over 32 keys, contraction HD=128 in 4 WMMA steps
    v8f s0 = {}, s1 = {};
    const __bf16* krow =
        Km + (size_t)(b * SD + kk + am) * DIMD + h * HDD + kh * 16;
#pragma unroll
    for (int c = 0; c < 4; ++c) {
      v16bf kf = *reinterpret_cast<const v16bf*>(krow + c * 32);
      s0 = wmma_bf16(qf[c].v, kf, s0);
    }
#pragma unroll
    for (int c = 0; c < 4; ++c) {
      v16bf kf = *reinterpret_cast<const v16bf*>(krow + (size_t)16 * DIMD + c * 32);
      s1 = wmma_bf16(qf[c].v, kf, s1);
    }

    float p0[8], p1[8];
#pragma unroll
    for (int r = 0; r < 8; ++r) {
      int qi = qt * 16 + r + 8 * kh;          // this lane's D row
      float a = s0[r] * scale;
      float c2 = s1[r] * scale;
      p0[r] = (kk + am      <= qi) ? a  : -1e30f;   // causal mask
      p1[r] = (kk + 16 + am <= qi) ? c2 : -1e30f;
    }

    // online softmax: each row's 16 columns live in one 16-lane half
#pragma unroll
    for (int r = 0; r < 8; ++r) {
      float tm = fmaxf(p0[r], p1[r]);
#pragma unroll
      for (int off = 8; off >= 1; off >>= 1)
        tm = fmaxf(tm, __shfl_xor(tm, off, 16));
      float mn = fmaxf(mrow[r], tm);
      float corr = __expf(mrow[r] - mn);
      mrow[r] = mn;
      lrow[r] *= corr;
#pragma unroll
      for (int nt = 0; nt < 8; ++nt) o[nt][r] *= corr;
      p0[r] = __expf(p0[r] - mn);
      p1[r] = __expf(p1[r] - mn);
      float rs = p0[r] + p1[r];
#pragma unroll
      for (int off = 8; off >= 1; off >>= 1)
        rs += __shfl_xor(rs, off, 16);
      lrow[r] += rs;
    }

    // D-fragment -> A-fragment transpose of P via wave-private LDS
#pragma unroll
    for (int r = 0; r < 8; ++r) {
      pw[(r + 8 * kh) * 32 + am]      = (__bf16)p0[r];
      pw[(r + 8 * kh) * 32 + 16 + am] = (__bf16)p1[r];
    }
    asm volatile("s_wait_dscnt 0" ::: "memory");  // wave-local LDS ordering
    AFrag pf;
    pf.h[0] = *reinterpret_cast<const v8bf*>(pw + am * 32 + kh * 16);
    pf.h[1] = *reinterpret_cast<const v8bf*>(pw + am * 32 + kh * 16 + 8);

    // PV: contraction over 32 keys; Vt rows give contiguous 32B B-fragments
#pragma unroll
    for (int nt = 0; nt < 8; ++nt) {
      const __bf16* vrow =
          Vt + (size_t)(h * HDD + nt * 16 + am) * MD + b * SD + kk + kh * 16;
      v16bf vf = *reinterpret_cast<const v16bf*>(vrow);
      o[nt] = wmma_bf16(pf.v, vf, o[nt]);
    }
  }

  // normalize and store (l > 0 guaranteed by diagonal)
#pragma unroll
  for (int r = 0; r < 8; ++r) {
    float inv = 1.f / lrow[r];
    int m = b * SD + qt * 16 + r + 8 * kh;
#pragma unroll
    for (int nt = 0; nt < 8; ++nt)
      O[(size_t)m * DIMD + h * HDD + nt * 16 + am] = (__bf16)(o[nt][r] * inv);
  }
}

// ---------------------------------------------------------------------------
// Host-side orchestration
// ---------------------------------------------------------------------------
extern "C" void kernel_launch(void* const* d_in, const int* in_sizes, int n_in,
                              void* d_out, int out_size, void* d_ws, size_t ws_size,
                              hipStream_t stream) {
  // inputs: x, start_pos, freqs_cis, mask, wq, wk, wv, wo
  const float* x  = (const float*)d_in[0];
  const float* fc = (const float*)d_in[2];
  const float* wq = (const float*)d_in[4];
  const float* wk = (const float*)d_in[5];
  const float* wv = (const float*)d_in[6];
  const float* wo = (const float*)d_in[7];
  float* out = (float*)d_out;

  char* ws = (char*)d_ws;
  const size_t XSZ = (size_t)MD * DIMD * sizeof(__bf16);    // 16 MB
  const size_t WSZ = (size_t)DIMD * DIMD * sizeof(__bf16);  //  8 MB
  __bf16* xb    = (__bf16*)(ws);
  __bf16* wqb   = (__bf16*)(ws + XSZ);
  __bf16* wkb   = (__bf16*)(ws + XSZ + 1 * WSZ);
  __bf16* wvb   = (__bf16*)(ws + XSZ + 2 * WSZ);
  __bf16* wob   = (__bf16*)(ws + XSZ + 3 * WSZ);
  __bf16* qb    = (__bf16*)(ws + 1 * XSZ + 4 * WSZ);
  __bf16* kb    = (__bf16*)(ws + 2 * XSZ + 4 * WSZ);
  __bf16* vtb   = (__bf16*)(ws + 3 * XSZ + 4 * WSZ);
  __bf16* attnb = xb;   // xb is dead after the V projection; reuse for attn out

  const size_t nx = (size_t)MD * DIMD;
  const size_t nw = (size_t)DIMD * DIMD;
  cast_f32_to_bf16<<<(unsigned)((nx + 255) / 256), 256, 0, stream>>>(x,  xb,  nx);
  cast_f32_to_bf16<<<(unsigned)((nw + 255) / 256), 256, 0, stream>>>(wq, wqb, nw);
  cast_f32_to_bf16<<<(unsigned)((nw + 255) / 256), 256, 0, stream>>>(wk, wkb, nw);
  cast_f32_to_bf16<<<(unsigned)((nw + 255) / 256), 256, 0, stream>>>(wv, wvb, nw);
  cast_f32_to_bf16<<<(unsigned)((nw + 255) / 256), 256, 0, stream>>>(wo, wob, nw);

  dim3 gemmGrid(DIMD / 64, MD / 128);   // (32, 32)
  gemm_bf16_tdm<0><<<gemmGrid, 256, 0, stream>>>(xb, wqb, qb,  MD, DIMD, DIMD);
  gemm_bf16_tdm<0><<<gemmGrid, 256, 0, stream>>>(xb, wkb, kb,  MD, DIMD, DIMD);
  gemm_bf16_tdm<1><<<gemmGrid, 256, 0, stream>>>(xb, wvb, vtb, MD, DIMD, DIMD);

  const size_t np = (size_t)MD * NHD * (HDD / 2);
  rope_kernel<<<(unsigned)((np + 255) / 256), 256, 0, stream>>>(qb, kb, fc);

  flash_attn_wmma<<<dim3(BD * NHD, SD / 128), 256, 0, stream>>>(qb, kb, vtb, attnb);

  gemm_bf16_tdm<2><<<gemmGrid, 256, 0, stream>>>(attnb, wob, (void*)out,
                                                 MD, DIMD, DIMD);
}
